// TransformerEncoderBlock_32435593019954
// MI455X (gfx1250) — matrix operand order repaired and checked by reading, not compiled, not run on hardware
//
#include <hip/hip_runtime.h>
#include <hip/hip_bf16.h>
#include <math.h>

// ---------------------------------------------------------------------------
// CDNA5 (gfx1250) wave32 WMMA f16 GEMM building blocks.
// D = A(16x32 f16) x B(32x16 f16) + C(16x16 f32), 2x2 tiles per wave (32x32).
// Operand layouts (ISA 7.12.2):
//   A frag: lane = 16*khalf + m ; elems 0..7 = A[m][khalf*8 + 0..7],
//                                 elems 8..15 = A[m][16 + khalf*8 + 0..7]
//   B frag: lane = 16*kgrp + n  ; elems j   = B[k][n], same k order as A elem j
// this code uses k = 16*khalf + j for A and B alike; with B TRANSPOSED ([N,K]) both frags are contiguous.
// ---------------------------------------------------------------------------

typedef __attribute__((ext_vector_type(16))) _Float16 v16h;
typedef __attribute__((ext_vector_type(8)))  _Float16 v8h;
typedef __attribute__((ext_vector_type(8)))  float    v8f;

#define DEVINL __device__ __forceinline__

DEVINL v8f wmma_f16(v16h a, v16h b, v8f c) {
    return __builtin_amdgcn_wmma_f32_16x16x32_f16(
        /*neg_a=*/false, a, /*neg_b=*/false, b,
        /*c_mod=*/(short)0, c, /*reuse_a=*/false, /*reuse_b=*/false);
}

// A: row-major [.. , lda] at tile origin (already offset to row0,k=0)
// Bt: row-major [.. , ldb] at tile origin (already offset to col0,k=0) (= B transposed)
DEVINL void gemm32x32(const _Float16* __restrict__ A, int lda,
                      const _Float16* __restrict__ Bt, int ldb,
                      int K, int lane, v8f c[2][2])
{
    v8f z = {};
    c[0][0] = z; c[0][1] = z; c[1][0] = z; c[1][1] = z;
    const int m  = lane & 15;
    const int kh = lane >> 4;
    const _Float16* a0p = A  + (size_t)m * lda + kh * 16;
    const _Float16* a1p = a0p + (size_t)16 * lda;
    const _Float16* b0p = Bt + (size_t)m * ldb + kh * 16;
    const _Float16* b1p = b0p + (size_t)16 * ldb;
    for (int k = 0; k < K; k += 32) {
        v16h a0 = *(const v16h*)(a0p + k);
        v16h a1 = *(const v16h*)(a1p + k);
        v16h b0 = *(const v16h*)(b0p + k);
        v16h b1 = *(const v16h*)(b1p + k);
        c[0][0] = wmma_f16(a0, b0, c[0][0]);
        c[0][1] = wmma_f16(a0, b1, c[0][1]);
        c[1][0] = wmma_f16(a1, b0, c[1][0]);
        c[1][1] = wmma_f16(a1, b1, c[1][1]);
    }
}

// C/D layout: lane l holds col = l&15 ; VGPR r holds row = r + 8*(l>>4).
#define EPI_ROW(row0, ti, r, lane) ((row0) + (ti)*16 + (((lane) >> 4) << 3) + (r))
#define EPI_COL(col0, tj, lane)    ((col0) + (tj)*16 + ((lane) & 15))

// monotonic float<->uint mapping for atomicMax on floats
DEVINL unsigned fmap(float f) {
    unsigned b = __float_as_uint(f);
    return b ^ (unsigned)(((int)b >> 31) | 0x80000000);
}
DEVINL float funmap(unsigned m) {
    unsigned b = (m & 0x80000000u) ? (m ^ 0x80000000u) : ~m;
    return __uint_as_float(b);
}

// ---------------------------------------------------------------------------
// small utility kernels
// ---------------------------------------------------------------------------

__global__ void k_fill_u32(unsigned* __restrict__ p, unsigned v, size_t n) {
    size_t i = (size_t)blockIdx.x * 256 + threadIdx.x;
    if (i < n) p[i] = v;
}

__global__ void k_cvt_f16(const float* __restrict__ in, _Float16* __restrict__ out, size_t n) {
    size_t i = (size_t)blockIdx.x * 256 + threadIdx.x;
    if (i < n) out[i] = (_Float16)in[i];
}

// in f32 [R,C] -> out f16 [C,R]  (R,C multiples of 32)
__global__ void k_transpose_cvt(const float* __restrict__ in, _Float16* __restrict__ out,
                                int R, int C) {
    __shared__ float tile[32][33];
    int c0 = blockIdx.x * 32, r0 = blockIdx.y * 32;
    int tx = threadIdx.x, ty = threadIdx.y;       // 32 x 8
    for (int i = 0; i < 32; i += 8)
        tile[ty + i][tx] = in[(size_t)(r0 + ty + i) * C + c0 + tx];
    __syncthreads();
    for (int i = 0; i < 32; i += 8)
        out[(size_t)(c0 + ty + i) * R + r0 + tx] = (_Float16)tile[tx][ty + i];
}

// h [B*N, E] f16 -> hT [(b*H+hh)*Dh + d][N] f16
__global__ void k_transpose_h(const _Float16* __restrict__ h, _Float16* __restrict__ hT,
                              size_t total, int N, int E, int H, int Dh) {
    size_t i = (size_t)blockIdx.x * 256 + threadIdx.x;
    if (i >= total) return;
    int e = (int)(i % E);
    size_t row = i / E;
    int b = (int)(row / N), n = (int)(row % N);
    int hh = e / Dh, d = e % Dh;
    hT[((size_t)(b * H + hh) * Dh + d) * N + n] = h[i];
}

// LayerNorm: one block per row (256 threads). Optionally emits
// diag[(b*H+h)*N + n] = dScale * sum(h_head^2)  (dScale = 0.5*Dh^-0.5).
__global__ void k_layernorm(const float* __restrict__ xin, const float* __restrict__ g,
                            const float* __restrict__ bta, _Float16* __restrict__ out,
                            float* __restrict__ diag, int E, int N, int H, int Dh,
                            float dScale) {
    int row = blockIdx.x;
    const float* xr = xin + (size_t)row * E;
    __shared__ float red[256];
    __shared__ float hd[16];
    int t = threadIdx.x;
    float s = 0.f;
    for (int e = t; e < E; e += 256) s += xr[e];
    red[t] = s; __syncthreads();
    for (int k = 128; k > 0; k >>= 1) { if (t < k) red[t] += red[t + k]; __syncthreads(); }
    float mean = red[0] / E;
    __syncthreads();
    float v = 0.f;
    for (int e = t; e < E; e += 256) { float d = xr[e] - mean; v += d * d; }
    red[t] = v; __syncthreads();
    for (int k = 128; k > 0; k >>= 1) { if (t < k) red[t] += red[t + k]; __syncthreads(); }
    float rstd = rsqrtf(red[0] / E + 1e-5f);
    if (diag && t < H) hd[t] = 0.f;
    __syncthreads();
    for (int e = t; e < E; e += 256) {
        float hv = (xr[e] - mean) * rstd * g[e] + bta[e];
        out[(size_t)row * E + e] = (_Float16)hv;
        if (diag) atomicAdd(&hd[e / Dh], hv * hv);
    }
    if (diag) {
        __syncthreads();
        if (t < H) {
            int b = row / N, n = row % N;
            diag[((size_t)(b * H + t)) * N + n] = hd[t] * dScale;
        }
    }
}

// ---------------------------------------------------------------------------
// Pass 1: u = uscale * (h_head @ proj^T); reduce row-max -> stabq, max -> stabk
// grid (M/64, N/64, B*H), block 128 (4 waves)
// ---------------------------------------------------------------------------
__global__ void k_stab(const _Float16* __restrict__ h, const _Float16* __restrict__ projh,
                       unsigned* __restrict__ stabq, unsigned* __restrict__ stabk,
                       int N, int E, int Dh, int M, int H, float uscale) {
    int lane = threadIdx.x & 31, w = threadIdx.x >> 5;
    int bh = blockIdx.z, b = bh / H, hh = bh % H;
    int row0 = blockIdx.y * 64 + (w >> 1) * 32;
    int col0 = blockIdx.x * 64 + (w & 1) * 32;
    const _Float16* A  = h + ((size_t)b * N + row0) * E + hh * Dh;
    const _Float16* Bt = projh + (size_t)col0 * Dh;
    v8f c[2][2];
    gemm32x32(A, E, Bt, Dh, Dh, lane, c);
    float gm = -3.4e38f;
    for (int ti = 0; ti < 2; ++ti) {
        for (int r = 0; r < 8; ++r) {
            float rm = fmaxf(c[ti][0][r], c[ti][1][r]) * uscale;
            for (int d2 = 8; d2 >= 1; d2 >>= 1)
                rm = fmaxf(rm, __shfl_xor(rm, d2, 32));   // reduce over 16 cols per half
            gm = fmaxf(gm, rm);
            if ((lane & 15) == 0) {
                int grow = EPI_ROW(row0, ti, r, lane);
                atomicMax(stabq + (size_t)bh * N + grow, fmap(rm));
            }
        }
    }
    gm = fmaxf(gm, __shfl_xor(gm, 16, 32));
    if (lane == 0) atomicMax(stabk + bh, fmap(gm));
}

// ---------------------------------------------------------------------------
// Pass 2: recompute u; q_p (row-major), k_p^T (transposed), k_sum (atomic).
// ---------------------------------------------------------------------------
__global__ void k_features(const _Float16* __restrict__ h, const _Float16* __restrict__ projh,
                           const float* __restrict__ diag, const unsigned* __restrict__ stabq,
                           const unsigned* __restrict__ stabk,
                           _Float16* __restrict__ qp, _Float16* __restrict__ kpT,
                           float* __restrict__ ksum,
                           int N, int E, int Dh, int M, int H,
                           float uscale, float ratio, float eps) {
    int lane = threadIdx.x & 31, w = threadIdx.x >> 5;
    int bh = blockIdx.z, b = bh / H, hh = bh % H;
    int row0 = blockIdx.y * 64 + (w >> 1) * 32;
    int col0 = blockIdx.x * 64 + (w & 1) * 32;
    const _Float16* A  = h + ((size_t)b * N + row0) * E + hh * Dh;
    const _Float16* Bt = projh + (size_t)col0 * Dh;
    v8f c[2][2];
    gemm32x32(A, E, Bt, Dh, Dh, lane, c);
    float sk = funmap(stabk[bh]);
    for (int tj = 0; tj < 2; ++tj) {
        int gcol = EPI_COL(col0, tj, lane);
        float colsum = 0.f;
        for (int ti = 0; ti < 2; ++ti) {
            for (int r = 0; r < 8; ++r) {
                int grow = EPI_ROW(row0, ti, r, lane);
                float u  = uscale * c[ti][tj][r];
                float ex = u - diag[(size_t)bh * N + grow];
                float sq = funmap(stabq[(size_t)bh * N + grow]);
                float qv = ratio * (expf(ex - sq) + eps);
                float kv = ratio * (expf(ex - sk) + eps);
                qp [(size_t)bh * N * M + (size_t)grow * M + gcol] = (_Float16)qv;
                kpT[(size_t)bh * M * N + (size_t)gcol * N + grow] = (_Float16)kv;
                colsum += kv;
            }
        }
        colsum += __shfl_xor(colsum, 16, 32);  // combine row halves
        if (lane < 16) atomicAdd(ksum + (size_t)bh * M + gcol, colsum);
    }
}

// ctx = k_p^T [M,N] @ qkv [N,Dh]; stored transposed: ctxt[bh][d][m]
// grid (1, M/64, B*H)
__global__ void k_ctx(const _Float16* __restrict__ kpT, const _Float16* __restrict__ hT,
                      _Float16* __restrict__ ctxt, int N, int M, int Dh) {
    int lane = threadIdx.x & 31, w = threadIdx.x >> 5;
    int bh = blockIdx.z;
    int row0 = blockIdx.y * 64 + (w >> 1) * 32;   // over M
    int col0 = (w & 1) * 32;                      // over Dh
    const _Float16* A  = kpT + (size_t)bh * M * N + (size_t)row0 * N;
    const _Float16* Bt = hT  + (size_t)bh * Dh * N + (size_t)col0 * N;
    v8f c[2][2];
    gemm32x32(A, N, Bt, N, N, lane, c);
    for (int ti = 0; ti < 2; ++ti)
        for (int tj = 0; tj < 2; ++tj)
            for (int r = 0; r < 8; ++r) {
                int grow = EPI_ROW(row0, ti, r, lane);
                int gcol = EPI_COL(col0, tj, lane);
                ctxt[((size_t)bh * Dh + gcol) * M + grow] = (_Float16)c[ti][tj][r];
            }
}

// dinv[bh*N+n] = 1 / dot(q_p[n,:], k_sum[bh,:])
__global__ void k_dinv(const _Float16* __restrict__ qp, const float* __restrict__ ksum,
                       float* __restrict__ dinv, size_t total, int N, int M) {
    size_t i = (size_t)blockIdx.x * 256 + threadIdx.x;
    if (i >= total) return;
    size_t bh = i / N;
    const _Float16* q = qp + i * M;
    const float*    s = ksum + bh * M;
    float acc = 0.f;
    for (int m = 0; m < M; ++m) acc += (float)q[m] * s[m];
    dinv[i] = 1.f / acc;
}

// attn = (q_p @ ctx) * dinv ; x1 = x + attn  (grid (1, N/64, B*H))
__global__ void k_attn(const _Float16* __restrict__ qp, const _Float16* __restrict__ ctxt,
                       const float* __restrict__ dinv, const float* __restrict__ x,
                       float* __restrict__ x1, int N, int E, int Dh, int M, int H) {
    int lane = threadIdx.x & 31, w = threadIdx.x >> 5;
    int bh = blockIdx.z, b = bh / H, hh = bh % H;
    int row0 = blockIdx.y * 64 + (w >> 1) * 32;   // over N
    int col0 = (w & 1) * 32;                      // over Dh
    const _Float16* A  = qp   + (size_t)bh * N * M + (size_t)row0 * M;
    const _Float16* Bt = ctxt + (size_t)bh * Dh * M + (size_t)col0 * M;
    v8f c[2][2];
    gemm32x32(A, M, Bt, M, M, lane, c);
    for (int ti = 0; ti < 2; ++ti)
        for (int tj = 0; tj < 2; ++tj)
            for (int r = 0; r < 8; ++r) {
                int grow = EPI_ROW(row0, ti, r, lane);
                int gcol = EPI_COL(col0, tj, lane);
                float di = dinv[(size_t)bh * N + grow];
                size_t idx = ((size_t)(b * N + grow)) * E + hh * Dh + gcol;
                x1[idx] = x[idx] + c[ti][tj][r] * di;
            }
}

// y1 = gelu(h2 @ W1 + b1)   grid (F/64, BN/64)
__global__ void k_mlp1(const _Float16* __restrict__ h2, const _Float16* __restrict__ W1t,
                       const float* __restrict__ b1, _Float16* __restrict__ y1,
                       int E, int F) {
    int lane = threadIdx.x & 31, w = threadIdx.x >> 5;
    int row0 = blockIdx.y * 64 + (w >> 1) * 32;
    int col0 = blockIdx.x * 64 + (w & 1) * 32;
    const _Float16* A  = h2 + (size_t)row0 * E;
    const _Float16* Bt = W1t + (size_t)col0 * E;
    v8f c[2][2];
    gemm32x32(A, E, Bt, E, E, lane, c);
    for (int ti = 0; ti < 2; ++ti)
        for (int tj = 0; tj < 2; ++tj)
            for (int r = 0; r < 8; ++r) {
                int grow = EPI_ROW(row0, ti, r, lane);
                int gcol = EPI_COL(col0, tj, lane);
                float t = c[ti][tj][r] + b1[gcol];
                float g = 0.5f * t * (1.f + erff(t * 0.70710678118654752f));
                y1[(size_t)grow * F + gcol] = (_Float16)g;
            }
}

// out = x1 + y1 @ W2 + b2   grid (E/64, BN/64)
__global__ void k_mlp2(const _Float16* __restrict__ y1, const _Float16* __restrict__ W2t,
                       const float* __restrict__ b2, const float* __restrict__ x1,
                       float* __restrict__ out, int E, int F) {
    int lane = threadIdx.x & 31, w = threadIdx.x >> 5;
    int row0 = blockIdx.y * 64 + (w >> 1) * 32;
    int col0 = blockIdx.x * 64 + (w & 1) * 32;
    const _Float16* A  = y1 + (size_t)row0 * F;
    const _Float16* Bt = W2t + (size_t)col0 * F;
    v8f c[2][2];
    gemm32x32(A, F, Bt, F, F, lane, c);
    for (int ti = 0; ti < 2; ++ti)
        for (int tj = 0; tj < 2; ++tj)
            for (int r = 0; r < 8; ++r) {
                int grow = EPI_ROW(row0, ti, r, lane);
                int gcol = EPI_COL(col0, tj, lane);
                size_t idx = (size_t)grow * E + gcol;
                out[idx] = x1[idx] + c[ti][tj][r] + b2[gcol];
            }
}

// ---------------------------------------------------------------------------

extern "C" void kernel_launch(void* const* d_in, const int* in_sizes, int n_in,
                              void* d_out, int out_size, void* d_ws, size_t ws_size,
                              hipStream_t stream) {
    const int B = 4, N = 2048, E = 768, H = 12, M = 256, F = 3072, Dh = 64;
    const int BH = B * H, BN = B * N;
    (void)in_sizes; (void)n_in; (void)out_size; (void)ws_size;

    const float* x    = (const float*)d_in[0];
    const float* proj = (const float*)d_in[1];
    const float* ln1g = (const float*)d_in[2];
    const float* ln1b = (const float*)d_in[3];
    const float* ln2g = (const float*)d_in[4];
    const float* ln2b = (const float*)d_in[5];
    const float* W1   = (const float*)d_in[6];
    const float* b1   = (const float*)d_in[7];
    const float* W2   = (const float*)d_in[8];
    const float* b2   = (const float*)d_in[9];
    float* out = (float*)d_out;

    char* wsb = (char*)d_ws;
    size_t off = 0;
    auto take = [&](size_t bytes) -> char* {
        char* p = wsb + off;
        off += (bytes + 255) & ~(size_t)255;
        return p;
    };
    _Float16* h     = (_Float16*)take((size_t)BN * E * 2);
    _Float16* hT    = (_Float16*)take((size_t)BH * Dh * N * 2);
    _Float16* projh = (_Float16*)take((size_t)M * Dh * 2);
    _Float16* W1t   = (_Float16*)take((size_t)F * E * 2);
    _Float16* W2t   = (_Float16*)take((size_t)E * F * 2);
    float*    diag  = (float*)   take((size_t)BH * N * 4);
    unsigned* stabq = (unsigned*)take((size_t)BH * N * 4);
    unsigned* stabk = (unsigned*)take((size_t)BH * 4);
    _Float16* qp    = (_Float16*)take((size_t)BH * N * M * 2);
    _Float16* kpT   = (_Float16*)take((size_t)BH * M * N * 2);
    float*    ksum  = (float*)   take((size_t)BH * M * 4);
    _Float16* ctxt  = (_Float16*)take((size_t)BH * Dh * M * 2);
    float*    dinv  = (float*)   take((size_t)BH * N * 4);
    float*    x1    = (float*)   take((size_t)BN * E * 4);
    _Float16* h2    = (_Float16*)take((size_t)BN * E * 2);
    _Float16* y1    = (_Float16*)take((size_t)BN * F * 2);

    const float uscale = powf((float)Dh, -0.25f);        // d^{-1/4}
    const float dScale = 0.5f / sqrtf((float)Dh);        // 0.5 * d^{-1/2}
    const float ratio  = rsqrtf((float)M);               // M^{-1/2}

    // weight / proj prep
    k_cvt_f16<<<(M * Dh + 255) / 256, 256, 0, stream>>>(proj, projh, (size_t)M * Dh);
    k_transpose_cvt<<<dim3(F / 32, E / 32), dim3(32, 8), 0, stream>>>(W1, W1t, E, F);
    k_transpose_cvt<<<dim3(E / 32, F / 32), dim3(32, 8), 0, stream>>>(W2, W2t, F, E);

    // zero accumulators (0u == fmap(-huge) for the max buffers, 0.0f for ksum)
    size_t nz1 = (size_t)BH * N;
    k_fill_u32<<<(unsigned)((nz1 + 255) / 256), 256, 0, stream>>>(stabq, 0u, nz1);
    k_fill_u32<<<1, 256, 0, stream>>>(stabk, 0u, (size_t)BH);
    k_fill_u32<<<(unsigned)(((size_t)BH * M + 255) / 256), 256, 0, stream>>>((unsigned*)ksum, 0u, (size_t)BH * M);

    // attention sublayer
    k_layernorm<<<BN, 256, 0, stream>>>(x, ln1g, ln1b, h, diag, E, N, H, Dh, dScale);
    k_stab<<<dim3(M / 64, N / 64, BH), 128, 0, stream>>>(h, projh, stabq, stabk, N, E, Dh, M, H, uscale);
    k_features<<<dim3(M / 64, N / 64, BH), 128, 0, stream>>>(h, projh, diag, stabq, stabk,
                                                             qp, kpT, ksum, N, E, Dh, M, H,
                                                             uscale, ratio, 1e-4f);
    k_transpose_h<<<(unsigned)(((size_t)BN * E + 255) / 256), 256, 0, stream>>>(h, hT, (size_t)BN * E, N, E, H, Dh);
    k_ctx<<<dim3(1, M / 64, BH), 128, 0, stream>>>(kpT, hT, ctxt, N, M, Dh);
    k_dinv<<<(unsigned)(((size_t)BH * N + 255) / 256), 256, 0, stream>>>(qp, ksum, dinv, (size_t)BH * N, N, M);
    k_attn<<<dim3(1, N / 64, BH), 128, 0, stream>>>(qp, ctxt, dinv, x, x1, N, E, Dh, M, H);

    // MLP sublayer
    k_layernorm<<<BN, 256, 0, stream>>>(x1, ln2g, ln2b, h2, nullptr, E, N, H, Dh, dScale);
    k_mlp1<<<dim3(F / 64, BN / 64), 128, 0, stream>>>(h2, W1t, b1, y1, E, F);
    k_mlp2<<<dim3(E / 64, BN / 64), 128, 0, stream>>>(y1, W2t, b2, x1, out, E, F);
}
